// Attention_STInf_5738076308226
// MI455X (gfx1250) — compile-verified, matrix operand order checked by reading, not run on hardware
//
#include <hip/hip_runtime.h>

#define BS   256
#define TM1  127
#define DD   128
#define DT   16
#define DB   32
#define DH   256
#define NH   4
#define DS   64
#define DHN  1024
#define KIN  160   // padded (DD+DT)=144 -> 160 ; (DD+DB)=160 exactly

typedef __attribute__((ext_vector_type(16))) __bf16 v16bf;
typedef __attribute__((ext_vector_type(8)))  float  v8f;
typedef __attribute__((ext_vector_type(4)))  int    v4i;

#if defined(__has_builtin)
#if __has_builtin(__builtin_amdgcn_global_load_async_to_lds_b128) && \
    __has_builtin(__builtin_amdgcn_s_wait_asynccnt)
#define HAVE_GFX1250_ASYNC 1
#endif
#endif

#ifdef HAVE_GFX1250_ASYNC
#define ASYNC_WAIT(n) __builtin_amdgcn_s_wait_asynccnt(n)
#else
#define ASYNC_WAIT(n)
#endif

__device__ __forceinline__ unsigned short f2bf(float f) {
    unsigned u = __float_as_uint(f);
    u += 0x7FFFu + ((u >> 16) & 1u);           // round-to-nearest-even
    return (unsigned short)(u >> 16);
}
__device__ __forceinline__ float bf2f(unsigned short h) {
    return __uint_as_float(((unsigned)h) << 16);
}

// copy one 16-byte chunk global -> LDS (async on gfx1250, sync fallback)
__device__ __forceinline__ void tile_copy16(unsigned short* dst_lds,
                                            const unsigned short* src_g) {
#ifdef HAVE_GFX1250_ASYNC
    __builtin_amdgcn_global_load_async_to_lds_b128(
        (v4i*)src_g, (v4i*)dst_lds, 0, 0);
#else
    *(uint4*)dst_lds = *(const uint4*)src_g;
#endif
}

// ---------------------------------------------------------------------------
// bf16 WMMA GEMM:  C[M,N] = act(A[M,K] @ Bp + bias)
// A: row-major bf16, row stride K (K mult of 32). Bp: pre-packed fragments:
// tile(kt,nt) -> 32 lanes x 16 bf16 contiguous.  B tiles for the block's
// 64-col group are staged in LDS with async double-buffering; all 8 waves of
// a block share the same ng (launcher guarantees waves % 8 == 0, MT % 8 == 0).
// All 4 B fragments are loaded from LDS before the 4 WMMAs so the matrix
// pipe issues back-to-back.  epi: 0=none 1=relu 2=softplus.
// ---------------------------------------------------------------------------
__global__ void wmma_gemm_bf16(const unsigned short* __restrict__ A,
                               const unsigned short* __restrict__ Bp,
                               const float* __restrict__ bias,
                               float* __restrict__ Cf, int ldcf,
                               unsigned short* __restrict__ Cb,
                               int M, int N, int K, int epi)
{
    __shared__ __align__(16) unsigned short Bs[2][4 * 512];   // 2 x 4KB
    const int tid  = threadIdx.x;
    const int lane = tid & 31;
    const int wave = blockIdx.x * 8 + (tid >> 5);
    const int MT = M >> 4;
    const int NT = N >> 4;
    const int KT = K >> 5;
    const int ng    = (blockIdx.x * 8) / MT;   // uniform across the block
    const int mtile = wave % MT;
    const int m0 = mtile << 4;

    const unsigned short* Bt = Bp + (size_t)(ng << 2) * 512;  // kt=0 tile group
    const size_t kstep = (size_t)NT * 512;

    // prologue: stage tile kt=0
    tile_copy16(&Bs[0][tid * 8], Bt + tid * 8);

    const unsigned short* Arow =
        A + (size_t)(m0 + (lane & 15)) * K + ((lane >> 4) & 1) * 8;

    v8f acc[4] = {};
    for (int kt = 0; kt < KT; ++kt) {
        const int cur = kt & 1;
        if (kt + 1 < KT)   // prefetch next tile into the other buffer
            tile_copy16(&Bs[cur ^ 1][tid * 8],
                        Bt + (size_t)(kt + 1) * kstep + tid * 8);

        union { uint4 u[2]; v16bf v; } af;
        const unsigned short* pa = Arow + kt * 32;
        af.u[0] = *(const uint4*)(pa);        // K = k0+sel*8 .. +7
        af.u[1] = *(const uint4*)(pa + 16);   // K = k0+16+sel*8 .. +7

        if (kt + 1 < KT) { ASYNC_WAIT(1); } else { ASYNC_WAIT(0); }
        __syncthreads();                       // current tile resident

        union { uint4 u[2]; v16bf v; } bu[4];
#pragma unroll
        for (int j = 0; j < 4; ++j) {          // one LDS clause, one wait
            const unsigned short* pb = &Bs[cur][j * 512 + lane * 16];
            bu[j].u[0] = *(const uint4*)(pb);
            bu[j].u[1] = *(const uint4*)(pb + 8);
        }
#pragma unroll
        for (int j = 0; j < 4; ++j) {          // back-to-back WMMA issue
            acc[j] = __builtin_amdgcn_wmma_f32_16x16x32_bf16(
                false, af.v, false, bu[j].v, (short)0, acc[j], false, false);
        }
        __syncthreads();                       // done reading before refill
    }

    const int rbase = m0 + ((lane >> 4) << 3);
#pragma unroll
    for (int j = 0; j < 4; ++j) {
        const int col = (((ng << 2) + j) << 4) + (lane & 15);
        const float bv = bias ? bias[col] : 0.0f;
#pragma unroll
        for (int r = 0; r < 8; ++r) {
            float v = acc[j][r] + bv;
            if (epi == 1)      v = fmaxf(v, 0.0f);
            else if (epi == 2) v = (v > 20.0f) ? v : logf(1.0f + expf(v));
            const int row = rbase + r;
            if (Cf) Cf[(size_t)row * ldcf + col] = v;
            if (Cb) Cb[(size_t)row * N + col] = f2bf(v);
        }
    }
}

// Pack weight W[Ksrc,N] f32 (zero-padded to Kpad) into B-fragment order, bf16.
__global__ void pack_w_kernel(const float* __restrict__ W,
                              unsigned short* __restrict__ Bp,
                              int Ksrc, int Kpad, int N)
{
    const int idx = blockIdx.x * blockDim.x + threadIdx.x;
    const int total = Kpad * N;
    if (idx >= total) return;
    const int e    = idx & 15;
    const int lane = (idx >> 4) & 31;
    const int tile = idx >> 9;
    const int NT = N >> 4;
    const int kt = tile / NT, nt = tile % NT;
    const int hi = (lane >> 4) & 1;
    const int kl = (e < 8) ? (e + hi * 8) : (16 + (e - 8) + hi * 8);
    const int k = kt * 32 + kl;
    const int n = nt * 16 + (lane & 15);
    const float v = (k < Ksrc) ? W[(size_t)k * N + n] : 0.0f;
    Bp[idx] = f2bf(v);
}

// inp_bf[b*127+t, 160] = [x[b,t+1,0:128] | a[b,t,0:16] | zeros]
__global__ void build_inp_kernel(const float* __restrict__ x,
                                 const float* __restrict__ a,
                                 unsigned short* __restrict__ inp)
{
    const int idx = blockIdx.x * blockDim.x + threadIdx.x;
    if (idx >= BS * TM1 * KIN) return;
    const int c = idx % KIN;
    const int row = idx / KIN;
    const int b = row / TM1, t = row % TM1;
    float v = 0.0f;
    if (c < DD)            v = x[((size_t)b * 128 + (t + 1)) * DD + c];
    else if (c < DD + DT)  v = a[((size_t)b * 128 + t) * DT + (c - DD)];
    inp[idx] = f2bf(v);
}

// xb_bf[b,160] = [x[b,0,:] | bvec[b,:]]
__global__ void build_xb_kernel(const float* __restrict__ x,
                                const float* __restrict__ bvec,
                                unsigned short* __restrict__ xb)
{
    const int idx = blockIdx.x * blockDim.x + threadIdx.x;
    if (idx >= BS * KIN) return;
    const int c = idx % KIN;
    const int b = idx / KIN;
    float v = (c < DD) ? x[(size_t)b * 128 * DD + c] : bvec[(size_t)b * DB + (c - DD)];
    xb[idx] = f2bf(v);
}

__global__ void build_mask_kernel(const float* __restrict__ m,
                                  float* __restrict__ maskw)
{
    const int idx = blockIdx.x * blockDim.x + threadIdx.x;
    if (idx >= BS * TM1) return;
    const int b = idx / TM1, t = idx % TM1;
    const float* p = m + ((size_t)b * 128 + (t + 1)) * DD;
    float s = 0.0f;
    for (int d = 0; d < DD; ++d) s += p[d];
    maskw[idx] = (s > 0.0f) ? 1.0f : 0.0f;
}

// Per-batch attention + gate:  ht = relu(0.5*(attn(key)+val)) -> bf16
// q/v streamed from (L2-resident) bf16 tensors, vectorized uint2 accesses.
__global__ void attn_h_kernel(const float* __restrict__ key,
                              const float* __restrict__ val,
                              const unsigned short* __restrict__ q_bf,
                              const unsigned short* __restrict__ v_bf,
                              const float* __restrict__ maskw,
                              unsigned short* __restrict__ ht_bf)
{
    __shared__ float key_s[DH];
    __shared__ float part[TM1 * 2][NH];   // split-d partial dots
    __shared__ float sc[TM1 * NH];
    __shared__ float inv_s[NH];
    const int b = blockIdx.x;
    const int tid = threadIdx.x;

    key_s[tid] = key[(size_t)b * DH + tid];       // blockDim == DH == 256
    __syncthreads();

    // scores: thread = (t, d-half); one uint2 = q[t, d, h=0..3]
    if (tid < 2 * TM1) {
        const int t = tid >> 1, half = tid & 1;
        const unsigned short* qp =
            q_bf + (((size_t)b * TM1 + t) << 10) + (half << 9);
        const float* kp = key_s + (half << 7);
        float s0 = 0.f, s1 = 0.f, s2 = 0.f, s3 = 0.f;
        for (int d = 0; d < 128; ++d) {
            const uint2 q4 = *(const uint2*)(qp + (d << 2));
            const float kd = kp[d];
            s0 += kd * bf2f((unsigned short)(q4.x & 0xffffu));
            s1 += kd * bf2f((unsigned short)(q4.x >> 16));
            s2 += kd * bf2f((unsigned short)(q4.y & 0xffffu));
            s3 += kd * bf2f((unsigned short)(q4.y >> 16));
        }
        part[tid][0] = s0; part[tid][1] = s1;
        part[tid][2] = s2; part[tid][3] = s3;
    }
    __syncthreads();

    if (tid < NH) {
        float mx = -1e30f;
        for (int t = 0; t < TM1; ++t) {
            float s = (part[2 * t][tid] + part[2 * t + 1][tid]) * 0.0625f;
            if (!(maskw[b * TM1 + t] > 0.0f)) s = -1e9f;
            sc[t * NH + tid] = s;
            mx = fmaxf(mx, s);
        }
        float sum = 0.0f;
        for (int t = 0; t < TM1; ++t) {
            const float e = expf(sc[t * NH + tid] - mx);
            sc[t * NH + tid] = e;
            sum += e;
        }
        inv_s[tid] = 1.0f / sum;
    }
    __syncthreads();

    // output: thread = d (all 4 heads); one uint2 of v per t, coalesced
    {
        const unsigned short* vp =
            v_bf + (((size_t)b * TM1) << 10) + (tid << 2);
        float a0 = 0.f, a1 = 0.f, a2 = 0.f, a3 = 0.f;
        for (int t = 0; t < TM1; ++t) {
            const float* p = &sc[t * NH];
            const uint2 v4 = *(const uint2*)(vp + ((size_t)t << 10));
            a0 += p[0] * bf2f((unsigned short)(v4.x & 0xffffu));
            a1 += p[1] * bf2f((unsigned short)(v4.x >> 16));
            a2 += p[2] * bf2f((unsigned short)(v4.y & 0xffffu));
            a3 += p[3] * bf2f((unsigned short)(v4.y >> 16));
        }
        const float4 vl = *(const float4*)(val + (size_t)b * DHN + (tid << 2));
        const float h0 = fmaxf(0.5f * (a0 * inv_s[0] + vl.x), 0.0f);
        const float h1 = fmaxf(0.5f * (a1 * inv_s[1] + vl.y), 0.0f);
        const float h2 = fmaxf(0.5f * (a2 * inv_s[2] + vl.z), 0.0f);
        const float h3 = fmaxf(0.5f * (a3 * inv_s[3] + vl.w), 0.0f);
        uint2 o;
        o.x = (unsigned)f2bf(h0) | ((unsigned)f2bf(h1) << 16);
        o.y = (unsigned)f2bf(h2) | ((unsigned)f2bf(h3) << 16);
        *(uint2*)(ht_bf + (size_t)b * DHN + (tid << 2)) = o;
    }
}

// z = mu + sg*eps[t] ; write Z slice and z_bf for next-step GEMMs
__global__ void zcombine_kernel(const float* __restrict__ eps,
                                float* __restrict__ outZ,
                                const float* __restrict__ outMU,
                                const float* __restrict__ outSG,
                                unsigned short* __restrict__ z_bf,
                                int t)
{
    const int idx = blockIdx.x * blockDim.x + threadIdx.x;
    if (idx >= BS * DS) return;
    const int b = idx >> 6, i = idx & 63;
    const size_t o = ((size_t)b * TM1 + t) * DS + i;
    const float z = outMU[o] + outSG[o] * eps[((size_t)t * BS + b) * DS + i];
    outZ[o] = z;
    z_bf[idx] = f2bf(z);
}

// ---------------------------------------------------------------------------

extern "C" void kernel_launch(void* const* d_in, const int* in_sizes, int n_in,
                              void* d_out, int out_size, void* d_ws, size_t ws_size,
                              hipStream_t stream)
{
    const float* x    = (const float*)d_in[0];
    const float* a    = (const float*)d_in[1];
    const float* m    = (const float*)d_in[2];
    const float* bvec = (const float*)d_in[3];
    const float* eps  = (const float*)d_in[4];
    const float* bk_w = (const float*)d_in[5];  const float* bk_b = (const float*)d_in[6];
    const float* bv_w = (const float*)d_in[7];  const float* bv_b = (const float*)d_in[8];
    const float* q_w  = (const float*)d_in[9];  const float* q_b  = (const float*)d_in[10];
    const float* v_w  = (const float*)d_in[11]; const float* v_b  = (const float*)d_in[12];
    const float* hk_w = (const float*)d_in[13]; const float* hk_b = (const float*)d_in[14];
    const float* hv_w = (const float*)d_in[15]; const float* hv_b = (const float*)d_in[16];
    const float* mu1_w= (const float*)d_in[17]; const float* mu1_b= (const float*)d_in[18];
    const float* sg1_w= (const float*)d_in[19]; const float* sg1_b= (const float*)d_in[20];
    const float* mut_w= (const float*)d_in[21]; const float* mut_b= (const float*)d_in[22];
    const float* sgt_w= (const float*)d_in[23]; const float* sgt_b= (const float*)d_in[24];

    float* outZ = (float*)d_out;
    const size_t ZSZ = (size_t)BS * TM1 * DS;
    float* outMU = outZ + ZSZ;
    float* outSG = outMU + ZSZ;

    char* ws = (char*)d_ws;
    size_t off = 0;
    auto carve = [&](size_t bytes) -> void* {
        void* p = ws + off;
        off += (bytes + 255) & ~(size_t)255;
        return p;
    };

    unsigned short* bkp  = (unsigned short*)carve((size_t)KIN * DH  * 2);
    unsigned short* bvp  = (unsigned short*)carve((size_t)KIN * DHN * 2);
    unsigned short* qp   = (unsigned short*)carve((size_t)KIN * DHN * 2);
    unsigned short* vp   = (unsigned short*)carve((size_t)KIN * DHN * 2);
    unsigned short* hkp  = (unsigned short*)carve((size_t)DS  * DH  * 2);
    unsigned short* hvp  = (unsigned short*)carve((size_t)DS  * DHN * 2);
    unsigned short* mu1p = (unsigned short*)carve((size_t)DHN * DS  * 2);
    unsigned short* sg1p = (unsigned short*)carve((size_t)DHN * DS  * 2);
    unsigned short* mutp = (unsigned short*)carve((size_t)DHN * DS  * 2);
    unsigned short* sgtp = (unsigned short*)carve((size_t)DHN * DS  * 2);
    unsigned short* inp_bf = (unsigned short*)carve((size_t)BS * TM1 * KIN * 2);
    unsigned short* xb_bf  = (unsigned short*)carve((size_t)BS * KIN * 2);
    float*          maskw  = (float*)carve((size_t)BS * TM1 * 4);
    unsigned short* q_bf   = (unsigned short*)carve((size_t)BS * TM1 * DHN * 2);
    unsigned short* v_bf   = (unsigned short*)carve((size_t)BS * TM1 * DHN * 2);
    float*          key_cur= (float*)carve((size_t)BS * DH * 4);
    float*          val_cur= (float*)carve((size_t)BS * DHN * 4);
    unsigned short* ht_bf  = (unsigned short*)carve((size_t)BS * DHN * 2);
    unsigned short* z_bf   = (unsigned short*)carve((size_t)BS * DS * 2);

    auto packw = [&](const float* W, unsigned short* P, int Ksrc, int Kpad, int N) {
        const int total = Kpad * N;
        pack_w_kernel<<<(total + 255) / 256, 256, 0, stream>>>(W, P, Ksrc, Kpad, N);
    };
    auto gemm = [&](const unsigned short* A, const unsigned short* Bp, const float* bias,
                    float* Cf, int ldcf, unsigned short* Cb, int M, int N, int K, int epi) {
        const int waves = (M >> 4) * (N >> 6);   // always a multiple of 8 here
        wmma_gemm_bf16<<<waves / 8, 256, 0, stream>>>(A, Bp, bias, Cf, ldcf, Cb, M, N, K, epi);
    };

    // ---- one-time preprocessing ----
    packw(bk_w,  bkp,  KIN, KIN, DH);
    packw(bv_w,  bvp,  KIN, KIN, DHN);
    packw(q_w,   qp,   DD + DT, KIN, DHN);   // zero-pad K 144 -> 160
    packw(v_w,   vp,   DD + DT, KIN, DHN);
    packw(hk_w,  hkp,  DS, DS, DH);
    packw(hv_w,  hvp,  DS, DS, DHN);
    packw(mu1_w, mu1p, DHN, DHN, DS);
    packw(sg1_w, sg1p, DHN, DHN, DS);
    packw(mut_w, mutp, DHN, DHN, DS);
    packw(sgt_w, sgtp, DHN, DHN, DS);

    {
        const int n1 = BS * TM1 * KIN;
        build_inp_kernel<<<(n1 + 255) / 256, 256, 0, stream>>>(x, a, inp_bf);
        const int n2 = BS * KIN;
        build_xb_kernel<<<(n2 + 255) / 256, 256, 0, stream>>>(x, bvec, xb_bf);
        const int n3 = BS * TM1;
        build_mask_kernel<<<(n3 + 255) / 256, 256, 0, stream>>>(m, maskw);
    }

    // big projections: q_inp = relu(inp@q_w+q_b), v_inp = inp@v_w+v_b  (bf16, L2-resident)
    gemm(inp_bf, qp, q_b, nullptr, 0, q_bf, BS * TM1, DHN, KIN, 1);
    gemm(inp_bf, vp, v_b, nullptr, 0, v_bf, BS * TM1, DHN, KIN, 0);

    // step-0 key/val from xb
    gemm(xb_bf, bkp, bk_b, key_cur, DH,  nullptr, BS, DH,  KIN, 1);
    gemm(xb_bf, bvp, bv_b, val_cur, DHN, nullptr, BS, DHN, KIN, 0);

    // ---- sequential recurrence ----
    for (int t = 0; t < TM1; ++t) {
        attn_h_kernel<<<BS, 256, 0, stream>>>(key_cur, val_cur, q_bf, v_bf, maskw, ht_bf);

        const unsigned short* mp = (t == 0) ? mu1p : mutp;
        const float*          mb = (t == 0) ? mu1_b : mut_b;
        const unsigned short* sp = (t == 0) ? sg1p : sgtp;
        const float*          sb = (t == 0) ? sg1_b : sgt_b;

        gemm(ht_bf, mp, mb, outMU + (size_t)t * DS, TM1 * DS, nullptr, BS, DS, DHN, 0);
        gemm(ht_bf, sp, sb, outSG + (size_t)t * DS, TM1 * DS, nullptr, BS, DS, DHN, 2);

        zcombine_kernel<<<(BS * DS) / 256, 256, 0, stream>>>(eps, outZ, outMU, outSG, z_bf, t);

        if (t + 1 < TM1) {
            gemm(z_bf, hkp, hk_b, key_cur, DH,  nullptr, BS, DH,  DS, 1);
            gemm(z_bf, hvp, hv_b, val_cur, DHN, nullptr, BS, DHN, DS, 0);
        }
    }
}
